// Refiner_38543036514695
// MI455X (gfx1250) — matrix operand order repaired and checked by reading, not compiled, not run on hardware
//
#include <hip/hip_runtime.h>
#include <hip/hip_bf16.h>

typedef __attribute__((ext_vector_type(16))) _Float16 v16h;
typedef __attribute__((ext_vector_type(8)))  float    v8f;

// ---------------- problem dimensions ----------------
#define B_   2
#define HF   1088
#define WF   1920
#define HH   544
#define WH   960
#define HQ   272
#define WQ   480
#define NQ   (HQ*WQ)          // 130560 quarter-res pixels per image
#define KSEL 5000             // top-K regions per image
#define NP   (B_*KSEL)        // 10000 patches
#define GP   4                // patches per workgroup
#define NGROUPS (NP/GP)       // 2500
#define EPS_ 1e-5f

// d_out layout (floats): pha_f, fgr_f, ref  (concat in return order)
#define OUT_PHA 0
#define OUT_FGR (B_*HF*WF)                 // 4,177,920
#define OUT_REF (OUT_FGR + B_*HF*WF*3)    // 16,711,680

// workspace layout (bytes)
#define WS_PATCH 0                          // int[NP]
#define WS_WP1   40960                      // [14][2][32][16] f16
#define WS_WP2   (WS_WP1 + 14*2*512*2)      // [9][1][32][16] f16
#define WS_WP3   (WS_WP2 + 9*512*2)
#define WS_WP4   (WS_WP3 + 9*512*2)         // [5][1][32][16] f16
#define WS_BIAS  (WS_WP4 + 5*512*2)         // 4 x float[32]

static __device__ __forceinline__ unsigned fkey(float x) {
  unsigned u = __float_as_uint(x);
  return (u & 0x80000000u) ? ~u : (u | 0x80000000u);
}

// Compile-time offset of a 16-element chunk of the padded im2col raster
// (CINP multiple of 16 => a chunk never crosses a 3x3-window segment).
// j16 = 16 * chunk_index.  Chunks past the real window clamp to 0 (their
// packed weights are zero, and offset 0 reads finite in-bounds data).
template<int CINP, int IS>
__host__ __device__ constexpr int chunk_off(int j16) {
  return (j16 < 9 * CINP)
             ? (((j16 / CINP) / 3) * IS + ((j16 / CINP) % 3)) * CINP + (j16 % CINP)
             : 0;
}

// -------------------------------------------------------------------------
// Weight pack: fold BN scale into weights, emit f16 B fragments.
// B layout (this code's contiguous k order): lane l holds column n=l&15; element e holds physical
// k = 32*s + e + 16*(l>>4).  klog = klp (no permutation), so the
// A side reads contiguous chunks.
// klog indexes the CHANNEL-PADDED im2col raster (CINP per tap).
// -------------------------------------------------------------------------
__global__ void pack_weights_kernel(
    const float* w1, const float* g1, const float* b1, const float* m1, const float* v1,
    const float* w2, const float* g2, const float* b2, const float* m2, const float* v2,
    const float* w3, const float* g3, const float* b3, const float* m3, const float* v3,
    const float* w4, const float* b4,
    _Float16* wp1, _Float16* wp2, _Float16* wp3, _Float16* wp4, float* bias)
{
  __shared__ float sc[4][32];
  int t = threadIdx.x;
  if (t < 32) {
    float s1 = 0.f, bb1 = 0.f;
    if (t < 24) { s1 = g1[t] * rsqrtf(v1[t] + EPS_); bb1 = b1[t] - m1[t] * s1; }
    sc[0][t] = s1; bias[t] = bb1;
    float s2 = 0.f, bb2 = 0.f;
    if (t < 16) { s2 = g2[t] * rsqrtf(v2[t] + EPS_); bb2 = b2[t] - m2[t] * s2; }
    sc[1][t] = s2; bias[32 + t] = bb2;
    float s3 = 0.f, bb3 = 0.f;
    if (t < 12) { s3 = g3[t] * rsqrtf(v3[t] + EPS_); bb3 = b3[t] - m3[t] * s3; }
    sc[2][t] = s3; bias[64 + t] = bb3;
    float s4 = 0.f, bb4 = 0.f;
    if (t < 4)  { s4 = 1.f; bb4 = b4[t]; }
    sc[3][t] = s4; bias[96 + t] = bb4;
  }
  __syncthreads();
  auto pack = [&](const float* w, _Float16* wp, int CIN, int CINP, int COUT,
                  int KS, int NT, const float* scale) {
    int total = KS * NT * 512;
    for (int i = t; i < total; i += blockDim.x) {
      int e    = i & 15;
      int lane = (i >> 4) & 31;
      int nt   = (i >> 9) % NT;
      int s    = (i >> 9) / NT;
      int klp  = e + ((lane >> 4) << 4);   // physical k within 32-step
      int klog = klp;
      int k    = s * 32 + klog;            // padded-raster logical k
      int kk   = k / CINP;                 // 3x3 tap index
      int c    = k - kk * CINP;            // padded channel
      int n    = nt * 16 + (lane & 15);
      float val = 0.f;
      if (kk < 9 && c < CIN && n < COUT) val = w[(kk * CIN + c) * COUT + n] * scale[n];
      wp[i] = (_Float16)val;
    }
  };
  pack(w1, wp1, 42, 48, 24, 14, 2, sc[0]);
  pack(w2, wp2, 24, 32, 16,  9, 1, sc[1]);
  pack(w3, wp3, 22, 32, 12,  9, 1, sc[2]);
  pack(w4, wp4, 12, 16,  4,  5, 1, sc[3]);
}

// -------------------------------------------------------------------------
// Deterministic radix-select top-K (per batch) + ref mask + patch list.
// Ties at the threshold are taken in ascending index order (stable top_k).
// -------------------------------------------------------------------------
__device__ __forceinline__ int block_scan_excl(int v, int tid, int* s, int* total) {
  s[tid] = v;
  __syncthreads();
  for (int off = 1; off < 1024; off <<= 1) {
    int tmp = (tid >= off) ? s[tid - off] : 0;
    __syncthreads();
    s[tid] += tmp;
    __syncthreads();
  }
  *total = s[1023];
  int incl = s[tid];
  __syncthreads();
  return incl - v;
}

__global__ __launch_bounds__(1024) void select_topk_kernel(
    const float* err, int* patches, float* ref_out)
{
  int bb  = blockIdx.x;
  int tid = threadIdx.x;
  const float* e = err + bb * NQ;
  __shared__ int hist[256];
  __shared__ unsigned s_prefix;
  __shared__ int s_rem;
  __shared__ int s_scan[1024];
  __shared__ int s_eqbase, s_selbase;

  unsigned prefix = 0; int remaining = KSEL;
  for (int shift = 24; shift >= 0; shift -= 8) {
    for (int i = tid; i < 256; i += 1024) hist[i] = 0;
    __syncthreads();
    for (int q = tid; q < NQ; q += 1024) {
      unsigned u = fkey(e[q]);
      bool cand = (shift == 24) || ((u >> (shift + 8)) == (prefix >> (shift + 8)));
      if (cand) atomicAdd(&hist[(u >> shift) & 255], 1);
    }
    __syncthreads();
    if (tid == 0) {
      int acc = 0, bsel = 0;
      for (int bn = 255; bn >= 0; bn--) {
        if (acc + hist[bn] >= remaining) { bsel = bn; break; }
        acc += hist[bn];
      }
      s_prefix = prefix | ((unsigned)bsel << shift);
      s_rem = remaining - acc;
    }
    __syncthreads();
    prefix = s_prefix; remaining = s_rem;
    __syncthreads();
  }
  const unsigned T = prefix;
  const int R = remaining;                  // #ties to include
  if (tid == 0) { s_eqbase = 0; s_selbase = bb * KSEL; }
  __syncthreads();
  for (int q0 = 0; q0 < NQ; q0 += 1024) {
    int q = q0 + tid;
    bool valid = q < NQ;
    float ev = valid ? e[q] : 0.f;
    unsigned u = valid ? fkey(ev) : 0u;
    int isgt = (valid && u > T)  ? 1 : 0;
    int iseq = (valid && u == T) ? 1 : 0;
    int eqtot, seltot;
    int eqpref = block_scan_excl(iseq, tid, s_scan, &eqtot);
    int sel = isgt | ((iseq && (s_eqbase + eqpref) < R) ? 1 : 0);
    int selpref = block_scan_excl(sel, tid, s_scan, &seltot);
    if (sel) patches[s_selbase + selpref] = bb * NQ + q;
    if (valid) ref_out[bb * NQ + q] = (sel && ev > 0.f) ? 1.0f : 0.0f;
    __syncthreads();
    if (tid == 0) { s_eqbase += eqtot; s_selbase += seltot; }
    __syncthreads();
  }
}

// -------------------------------------------------------------------------
// 4x bilinear upsample of pha/fgr to full res (bandwidth-bound, ~67MB writes)
// -------------------------------------------------------------------------
__global__ __launch_bounds__(256) void upsample_kernel(
    const float* __restrict__ pha, const float* __restrict__ fgr, float* __restrict__ out)
{
  int idx = blockIdx.x * blockDim.x + threadIdx.x;
  if (idx >= B_ * HF * WF) return;
  int x = idx % WF;
  int y = (idx / WF) % HF;
  int bb = idx / (WF * HF);
  float fy = y * 0.25f - 0.375f;
  float fx = x * 0.25f - 0.375f;
  int y0 = (int)floorf(fy), x0 = (int)floorf(fx);
  float wy = fy - (float)y0, wx = fx - (float)x0;
  int y0c = max(y0, 0), y1c = min(y0 + 1, HQ - 1);
  int x0c = max(x0, 0), x1c = min(x0 + 1, WQ - 1);
  float w00 = (1.f - wy) * (1.f - wx), w01 = (1.f - wy) * wx;
  float w10 = wy * (1.f - wx),         w11 = wy * wx;
  int i00 = (bb * HQ + y0c) * WQ + x0c;
  int i01 = (bb * HQ + y0c) * WQ + x1c;
  int i10 = (bb * HQ + y1c) * WQ + x0c;
  int i11 = (bb * HQ + y1c) * WQ + x1c;
  out[OUT_PHA + idx] = pha[i00] * w00 + pha[i01] * w01 + pha[i10] * w10 + pha[i11] * w11;
  #pragma unroll
  for (int ch = 0; ch < 3; ch++) {
    out[OUT_FGR + idx * 3 + ch] =
        fgr[i00 * 3 + ch] * w00 + fgr[i01 * 3 + ch] * w01 +
        fgr[i10 * 3 + ch] * w10 + fgr[i11 * 3 + ch] * w11;
  }
}

// -------------------------------------------------------------------------
// Implicit-GEMM 3x3 conv on WMMA f32_16x16x32_f16, channel-padded inputs.
// A fragment = one contiguous 32B-aligned 16-element LDS chunk per lane:
//   chunk index j = 2*s + half, base offset = compile-time chunk_off(16*j),
//   resolved with a single v_cndmask -> two ds_load_b128.
// s is the outer loop so each A fragment is loaded exactly once per tile.
// -------------------------------------------------------------------------
template<int IS, int OS, int CINP, int KS, int NT, int STN, bool UP2, int OST,
         typename OutT, bool RELU>
__device__ __forceinline__ void conv3x3_wmma(
    const _Float16* __restrict__ in, OutT* __restrict__ out,
    const _Float16* __restrict__ wfrag, const float* __restrict__ bias, int tid)
{
  constexpr int MT = (GP * OS * OS) / 16;
  const int wave = tid >> 5;
  const int lane = tid & 31;
  const bool h1  = ((lane >> 4) != 0);
  const int l16  = lane & 15;
  for (int tile = wave; tile < MT; tile += 4) {
    // A-row of this lane: rows m = tile*16 + (lane&15) (both halves)
    int m   = tile * 16 + l16;
    int p   = m / (OS * OS);
    int pix = m - p * (OS * OS);
    int oy  = pix / OS;
    int ox  = pix - oy * OS;
    const _Float16* rowp = in + ((p * IS + oy) * IS + ox) * CINP;
    // D rows of this lane: m = tile*16 + half*8 + r
    int mm0 = tile * 16 + (h1 ? 8 : 0);
    int pp0 = mm0 / (OS * OS);
    int px0 = mm0 - pp0 * (OS * OS);
    v8f acc[NT];
    #pragma unroll
    for (int t2 = 0; t2 < NT; t2++)
      #pragma unroll
      for (int i = 0; i < 8; i++) acc[t2][i] = 0.f;
    #pragma unroll
    for (int s = 0; s < KS; s++) {
      const int c0 = chunk_off<CINP, IS>((2 * s) * 16);
      const int c1 = chunk_off<CINP, IS>((2 * s + 1) * 16);
      v16h a = *(const v16h*)(rowp + (h1 ? c1 : c0));
      #pragma unroll
      for (int t2 = 0; t2 < NT; t2++) {
        v16h bf = *(const v16h*)(wfrag + (size_t)((s * NT + t2) * 32 + lane) * 16);
        acc[t2] = __builtin_amdgcn_wmma_f32_16x16x32_f16(
            false, a, false, bf, (short)0, acc[t2], false, false);
      }
    }
    #pragma unroll
    for (int t2 = 0; t2 < NT; t2++) {
      int n = t2 * 16 + l16;
      if (n < STN) {
        float bv = bias[n];
        int p2 = pp0, x2 = px0;
        #pragma unroll
        for (int r = 0; r < 8; r++) {
          float vv = acc[t2][r] + bv;
          if (RELU) vv = fmaxf(vv, 0.f);
          if (!UP2) {
            out[(p2 * (OS * OS) + x2) * OST + n] = (OutT)vv;
          } else {
            int oyy = x2 / OS, oxx = x2 - (x2 / OS) * OS;  // OS=4 -> shifts
            #pragma unroll
            for (int a2 = 0; a2 < 2; a2++)
              #pragma unroll
              for (int b2 = 0; b2 < 2; b2++)
                out[((p2 * (2 * OS) + 2 * oyy + a2) * (2 * OS) + 2 * oxx + b2) * OST + n] =
                    (OutT)vv;
          }
          x2++;
          if (x2 == OS * OS) { x2 = 0; p2++; }
        }
      }
    }
  }
}

// -------------------------------------------------------------------------
// Patch refinement: gather -> conv1 -> conv2(+nearest2x) -> conv3 -> conv4 ->
// scatter 4x4 blocks into full-res outputs.  GP patches per 128-thread block.
// All LDS tensors channel-padded to multiples of 16 (pads zeroed).
// __launch_bounds__(128,4): cap at 256 VGPRs (no MSB mode), 4 waves/SIMD.
// -------------------------------------------------------------------------
__global__ __launch_bounds__(128, 4) void refine_patches_kernel(
    const float* __restrict__ src, const float* __restrict__ bgr,
    const float* __restrict__ pha, const float* __restrict__ fgr,
    const float* __restrict__ hid, const int* __restrict__ patches,
    const _Float16* __restrict__ wp1, const _Float16* __restrict__ wp2,
    const _Float16* __restrict__ wp3, const _Float16* __restrict__ wp4,
    const float* __restrict__ bias, float* __restrict__ out)
{
  __shared__ _Float16 s_in1[GP][8][8][48];   // [hid32, pha, fgr3, src3, bgr3, pad6]
  __shared__ _Float16 s_in3[GP][8][8][32];   // [conv2-up16, y2(6), pad10]
  __shared__ _Float16 s_t1 [GP][6][6][32];   // conv1 out (24 + zero pad8)
  __shared__ _Float16 s_t3 [GP][6][6][16];   // conv3 out (12 + zero pad4)
  __shared__ float    s_t4 [GP][4][4][4];
  __shared__ int s_pb[GP], s_iy[GP], s_ix[GP];

  int tid = threadIdx.x;
  if (tid < GP) {
    int id = patches[blockIdx.x * GP + tid];
    int bb = id / NQ;
    int q  = id - bb * NQ;
    s_pb[tid] = bb;
    s_iy[tid] = q / WQ;
    s_ix[tid] = q - (q / WQ) * WQ;
  }
  __syncthreads();

  // ---- Stage A: gather patch inputs into LDS (f16) ----
  for (int pos = tid; pos < GP * 64; pos += 128) {
    int p  = pos >> 6;
    int rr = (pos >> 3) & 7;
    int cc = pos & 7;
    int bb = s_pb[p];
    int r = s_iy[p] * 2 - 3 + rr;          // half-res window row
    int c = s_ix[p] * 2 - 3 + cc;
    _Float16* dst = &s_in1[p][rr][cc][0];
    if (r >= 0 && r < HH && c >= 0 && c < WH) {
      // bilinear 2x upsample of quarter-res x = [hid,pha,fgr]
      float fy = r * 0.5f - 0.25f, fx = c * 0.5f - 0.25f;
      int y0 = (int)floorf(fy), x0 = (int)floorf(fx);
      float wy = fy - (float)y0, wx = fx - (float)x0;
      int y0c = max(y0, 0), y1c = min(y0 + 1, HQ - 1);
      int x0c = max(x0, 0), x1c = min(x0 + 1, WQ - 1);
      float w00 = (1.f - wy) * (1.f - wx), w01 = (1.f - wy) * wx;
      float w10 = wy * (1.f - wx),         w11 = wy * wx;
      int i00 = (bb * HQ + y0c) * WQ + x0c;
      int i01 = (bb * HQ + y0c) * WQ + x1c;
      int i10 = (bb * HQ + y1c) * WQ + x0c;
      int i11 = (bb * HQ + y1c) * WQ + x1c;
      #pragma unroll 4
      for (int ch = 0; ch < 32; ch++)
        dst[ch] = (_Float16)(hid[i00 * 32 + ch] * w00 + hid[i01 * 32 + ch] * w01 +
                             hid[i10 * 32 + ch] * w10 + hid[i11 * 32 + ch] * w11);
      dst[32] = (_Float16)(pha[i00] * w00 + pha[i01] * w01 + pha[i10] * w10 + pha[i11] * w11);
      #pragma unroll
      for (int ch = 0; ch < 3; ch++)
        dst[33 + ch] = (_Float16)(fgr[i00 * 3 + ch] * w00 + fgr[i01 * 3 + ch] * w01 +
                                  fgr[i10 * 3 + ch] * w10 + fgr[i11 * 3 + ch] * w11);
      // 2x downsample (2x2 mean) of full-res src/bgr
      int yf = 2 * r, xf = 2 * c;
      int j00 = ((bb * HF + yf) * WF + xf) * 3;
      int j01 = j00 + 3;
      int j10 = j00 + WF * 3;
      int j11 = j10 + 3;
      #pragma unroll
      for (int ch = 0; ch < 3; ch++) {
        dst[36 + ch] = (_Float16)(0.25f * (src[j00 + ch] + src[j01 + ch] +
                                           src[j10 + ch] + src[j11 + ch]));
        dst[39 + ch] = (_Float16)(0.25f * (bgr[j00 + ch] + bgr[j01 + ch] +
                                           bgr[j10 + ch] + bgr[j11 + ch]));
      }
    } else {
      #pragma unroll 6
      for (int ch = 0; ch < 42; ch++) dst[ch] = (_Float16)0.f;
    }
    #pragma unroll
    for (int ch = 42; ch < 48; ch++) dst[ch] = (_Float16)0.f;  // channel pad
    // y2: full-res crop (size 4, pad 2) -> s_in3 channels 16..21 (+ pad 22..31)
    int r2 = s_iy[p] * 4 - 2 + rr;
    int c2 = s_ix[p] * 4 - 2 + cc;
    _Float16* d3 = &s_in3[p][rr][cc][16];
    if (r2 >= 0 && r2 < HF && c2 >= 0 && c2 < WF) {
      int ib = ((bb * HF + r2) * WF + c2) * 3;
      #pragma unroll
      for (int ch = 0; ch < 3; ch++) {
        d3[ch]     = (_Float16)src[ib + ch];
        d3[3 + ch] = (_Float16)bgr[ib + ch];
      }
    } else {
      #pragma unroll
      for (int ch = 0; ch < 6; ch++) d3[ch] = (_Float16)0.f;
    }
    #pragma unroll
    for (int ch = 6; ch < 16; ch++) d3[ch] = (_Float16)0.f;    // channel pad
  }
  __syncthreads();

  // conv1: 8x8x48p -> 6x6x32p (BN+ReLU folded; n>=24 stores exact zeros)
  conv3x3_wmma<8, 6, 48, 14, 2, 32, false, 32, _Float16, true>(
      &s_in1[0][0][0][0], &s_t1[0][0][0][0], wp1, bias + 0, tid);
  __syncthreads();
  // conv2: 6x6x32p -> 4x4x16, nearest-2x into s_in3[..][0..15]
  conv3x3_wmma<6, 4, 32, 9, 1, 16, true, 32, _Float16, true>(
      &s_t1[0][0][0][0], &s_in3[0][0][0][0], wp2, bias + 32, tid);
  __syncthreads();
  // conv3: 8x8x32p -> 6x6x16p (n>=12 stores exact zeros)
  conv3x3_wmma<8, 6, 32, 9, 1, 16, false, 16, _Float16, true>(
      &s_in3[0][0][0][0], &s_t3[0][0][0][0], wp3, bias + 64, tid);
  __syncthreads();
  // conv4: 6x6x16p -> 4x4x4 (no ReLU, f32 out, only n<4 stored)
  conv3x3_wmma<6, 4, 16, 5, 1, 4, false, 4, float, false>(
      &s_t3[0][0][0][0], &s_t4[0][0][0][0], wp4, bias + 96, tid);
  __syncthreads();

  // scatter refined 4x4 blocks into pha_f / fgr_f
  for (int pos = tid; pos < GP * 16; pos += 128) {
    int p = pos >> 4;
    int r = (pos >> 2) & 3;
    int c = pos & 3;
    int bb = s_pb[p];
    int y = s_iy[p] * 4 + r;
    int x = s_ix[p] * 4 + c;
    int pi = (bb * HF + y) * WF + x;
    out[OUT_PHA + pi]         = s_t4[p][r][c][0];
    out[OUT_FGR + pi * 3 + 0] = s_t4[p][r][c][1];
    out[OUT_FGR + pi * 3 + 1] = s_t4[p][r][c][2];
    out[OUT_FGR + pi * 3 + 2] = s_t4[p][r][c][3];
  }
}

// -------------------------------------------------------------------------
extern "C" void kernel_launch(void* const* d_in, const int* in_sizes, int n_in,
                              void* d_out, int out_size, void* d_ws, size_t ws_size,
                              hipStream_t stream) {
  const float* src = (const float*)d_in[0];
  const float* bgr = (const float*)d_in[1];
  const float* pha = (const float*)d_in[2];
  const float* fgr = (const float*)d_in[3];
  const float* err = (const float*)d_in[4];
  const float* hid = (const float*)d_in[5];
  const float* w1 = (const float*)d_in[6];
  const float* g1 = (const float*)d_in[7];
  const float* b1 = (const float*)d_in[8];
  const float* m1 = (const float*)d_in[9];
  const float* v1 = (const float*)d_in[10];
  const float* w2 = (const float*)d_in[11];
  const float* g2 = (const float*)d_in[12];
  const float* b2 = (const float*)d_in[13];
  const float* m2 = (const float*)d_in[14];
  const float* v2 = (const float*)d_in[15];
  const float* w3 = (const float*)d_in[16];
  const float* g3 = (const float*)d_in[17];
  const float* b3 = (const float*)d_in[18];
  const float* m3 = (const float*)d_in[19];
  const float* v3 = (const float*)d_in[20];
  const float* w4 = (const float*)d_in[21];
  const float* b4 = (const float*)d_in[22];
  float* out = (float*)d_out;

  int*      patches = (int*)((char*)d_ws + WS_PATCH);
  _Float16* wp1     = (_Float16*)((char*)d_ws + WS_WP1);
  _Float16* wp2     = (_Float16*)((char*)d_ws + WS_WP2);
  _Float16* wp3     = (_Float16*)((char*)d_ws + WS_WP3);
  _Float16* wp4     = (_Float16*)((char*)d_ws + WS_WP4);
  float*    bias    = (float*)((char*)d_ws + WS_BIAS);

  pack_weights_kernel<<<1, 256, 0, stream>>>(
      w1, g1, b1, m1, v1, w2, g2, b2, m2, v2, w3, g3, b3, m3, v3, w4, b4,
      wp1, wp2, wp3, wp4, bias);
  select_topk_kernel<<<B_, 1024, 0, stream>>>(err, patches, out + OUT_REF);
  upsample_kernel<<<(B_ * HF * WF + 255) / 256, 256, 0, stream>>>(pha, fgr, out);
  refine_patches_kernel<<<NGROUPS, 128, 0, stream>>>(
      src, bgr, pha, fgr, hid, patches, wp1, wp2, wp3, wp4, bias, out);
}
